// RelationshipHead_25262997635259
// MI455X (gfx1250) — matrix operand order repaired and checked by reading, not compiled, not run on hardware
//
#include <hip/hip_runtime.h>

// ---------------------------------------------------------------------------
// RelationshipHead for MI455X (gfx1250, wave32, WMMA f16 16x16x32)
//   - tile staging via GLOBAL_LOAD_ASYNC_TO_LDS_B128 (ASYNCcnt), 2-deep
//     prefetch, 4-buffer LDS rotation, one barrier per K-stage
//   - v_wmma_f32_16x16x32_f16 compute, 8 accumulators / wave (64x32 tile)
// Pipeline:
//   1) f32->f16 converts
//   2) QKV GEMM on the 1600 unique visual rows (dedup of pair gather)
//   3) per-pair 2x2 attention -> O f16 [2P,512]
//   4) out-proj GEMM -> comb[:, q*512:+512] (interleaved row map)
//   5) spatial GEMM  -> comb[:, 1024:1536]
//   6) fc1 +relu -> h1 ; 7) fc2 +relu -> h2 ; 8) fc3 +clip -> d_out
// ---------------------------------------------------------------------------

typedef _Float16 f16;
typedef __attribute__((ext_vector_type(16))) _Float16 v16h;
typedef __attribute__((ext_vector_type(8)))  _Float16 v8h;
typedef __attribute__((ext_vector_type(8)))  float    v8f;

#define DIM_B   16
#define DIM_N   100
#define DIM_M   4096
#define DIM_E   512
#define DIM_H   8
#define DIM_HD  64
#define DIM_3E  1536
#define DIM_HID 1024
#define DIM_H2  512
#define DIM_REL 50
#define NPAIRS  (DIM_B * DIM_M)      /* 65536 */
#define NROWS   (DIM_B * DIM_N)      /* 1600 unique visual rows */

// GEMM tiling
#define BM 128
#define BN 128
#define BK 64
#define LDSK 72      /* f16 stride: 144B rows, 16B aligned */
#define NBUF 4       /* LDS stage buffers, 2-deep async prefetch */

static inline int ceil_div_i(int a, int b) { return (a + b - 1) / b; }

// ---------------------------------------------------------------------------
// f32 -> f16 elementwise convert (8 elems / thread, vectorized)
// ---------------------------------------------------------------------------
__global__ __launch_bounds__(256) void cvt_f32_f16_kernel(
    const float* __restrict__ s, f16* __restrict__ d, int n)
{
    int i = blockIdx.x * blockDim.x + threadIdx.x;
    int base = i * 8;
    if (base + 8 <= n) {
        const float4* p = (const float4*)(s + base);
        float4 a = p[0], b = p[1];
        v8h o;
        o[0] = (_Float16)a.x; o[1] = (_Float16)a.y;
        o[2] = (_Float16)a.z; o[3] = (_Float16)a.w;
        o[4] = (_Float16)b.x; o[5] = (_Float16)b.y;
        o[6] = (_Float16)b.z; o[7] = (_Float16)b.w;
        *(v8h*)(d + base) = o;
    } else {
        for (int j = base; j < n; ++j) d[j] = (_Float16)s[j];
    }
}

// ---------------------------------------------------------------------------
// Async tile mover: 128x64 f16 tile, 256 threads x 4 chunks of 16B each.
// Writes LDS directly via GLOBAL_LOAD_ASYNC_TO_LDS_B128 (ASYNCcnt).
// LDS operand = low 32 bits of the flat shared-pointer (LDS aperture).
// Row index clamped so every issued load stays in-bounds.
// ---------------------------------------------------------------------------
__device__ __forceinline__ void async_tile_issue(
    const f16* __restrict__ G, int rows, int ld, int rowBase, int kOff,
    f16 (* __restrict__ L)[LDSK], int tid)
{
#pragma unroll
    for (int i = 0; i < 4; ++i) {
        int chunk = tid + i * 256;          // 1024 chunks of 8 f16
        int row   = chunk >> 3;
        int c     = (chunk & 7) * 8;
        int g     = rowBase + row;
        if (g >= rows) g = rows - 1;        // clamp: stays in-bounds, masked later
        unsigned long long gaddr =
            (unsigned long long)(const void*)(G + (size_t)g * ld + kOff + c);
        unsigned lds_addr = (unsigned)(unsigned long long)(const void*)&L[row][c];
        asm volatile("global_load_async_to_lds_b128 %0, %1, off"
                     :: "v"(lds_addr), "v"(gaddr) : "memory");
    }
}

__device__ __forceinline__ void wait_async_le16() {
    asm volatile("s_wait_asynccnt 0x10" ::: "memory");
}
__device__ __forceinline__ void wait_async_le8() {
    asm volatile("s_wait_asynccnt 0x8" ::: "memory");
}
__device__ __forceinline__ void wait_async_le0() {
    asm volatile("s_wait_asynccnt 0x0" ::: "memory");
}

// A-matrix fragment (16x32 f16), documented layout:
// lanes 0-15: VGPR0-3 = K 0..7,  VGPR4-7 = K 16..23
// lanes 16-31: VGPR0-3 = K 8..15, VGPR4-7 = K 24..31
__device__ __forceinline__ v16h frag_a(
    const f16 (* __restrict__ L)[LDSK], int row, int kk, int lh)
{
    const v8h lo = *(const v8h*)&L[row][kk + 8 * lh];
    const v8h hi = *(const v8h*)&L[row][kk + 16 + 8 * lh];
    v16h t;
#pragma unroll
    for (int j = 0; j < 8; ++j) { t[j] = lo[j]; t[j + 8] = hi[j]; }
    return t;
}

// B-matrix fragment (32x16 f16): lane = column, lanes 0-15 hold K 0..7,16..23,
// lanes 16-31 hold K 8..15,24..31 (same k order as A fragment)
__device__ __forceinline__ v16h frag_b(
    const f16 (* __restrict__ L)[LDSK], int row, int kk, int lh)
{
    const v8h lo = *(const v8h*)&L[row][kk + 8 * lh];
    const v8h hi = *(const v8h*)&L[row][kk + 16 + 8 * lh];
    v16h t;
#pragma unroll
    for (int j = 0; j < 8; ++j) { t[j] = lo[j]; t[j + 8] = hi[j]; }
    return t;
}

// ---------------------------------------------------------------------------
// Generic WMMA GEMM: C[M,N] = A[M,K] @ W[N,K]^T (+bias) (+relu) (+clip)
// ROWMAP 0: out offset = m*pitch + n
// ROWMAP 1: out offset = (m>>1)*pitch + (m&1)*512 + n   (att -> comb)
// ---------------------------------------------------------------------------
template <bool RELU, bool OUT_F16, int ROWMAP, bool CLIP>
__global__ __launch_bounds__(256) void gemm_wmma_kernel(
    const f16* __restrict__ A, const f16* __restrict__ W,
    const float* __restrict__ bias,
    float* __restrict__ outF, f16* __restrict__ outH,
    int M, int N, int K, int out_pitch)
{
    __shared__ f16 As[NBUF][BM][LDSK];
    __shared__ f16 Ws[NBUF][BN][LDSK];

    const int tid  = threadIdx.x;
    const int lane = tid & 31;
    const int wave = tid >> 5;
    const int wm   = wave >> 2;      // 0..1  (64 rows each)
    const int wn   = wave & 3;       // 0..3  (32 cols each)
    const int lr   = lane & 15;
    const int lh   = lane >> 4;
    const int mBase = blockIdx.x * BM;
    const int nBase = blockIdx.y * BN;

    v8f acc[4][2];
#pragma unroll
    for (int i = 0; i < 4; ++i)
#pragma unroll
        for (int j = 0; j < 2; ++j)
            acc[i][j] = (v8f){0.f, 0.f, 0.f, 0.f, 0.f, 0.f, 0.f, 0.f};

    const int S = K / BK;

    // 2-deep async prefetch into 4-buffer rotation
    async_tile_issue(A, M, K, mBase, 0, As[0], tid);
    async_tile_issue(W, N, K, nBase, 0, Ws[0], tid);
    if (S > 1) {
        async_tile_issue(A, M, K, mBase, BK, As[1], tid);
        async_tile_issue(W, N, K, nBase, BK, Ws[1], tid);
    }

    for (int s = 0; s < S; ++s) {
        const int cur = s & (NBUF - 1);
        if (s + 2 < S) {
            const int nxt = (s + 2) & (NBUF - 1);
            async_tile_issue(A, M, K, mBase, (s + 2) * BK, As[nxt], tid);
            async_tile_issue(W, N, K, nBase, (s + 2) * BK, Ws[nxt], tid);
        }
        // retire stage s (8 async loads/thread/tile-pair, in-order completion)
        const int rem = S - 1 - s;
        if (rem >= 2)      wait_async_le16();
        else if (rem == 1) wait_async_le8();
        else               wait_async_le0();
        __syncthreads();

#pragma unroll
        for (int kk = 0; kk < BK; kk += 32) {
            v16h af[4], bf[2];
#pragma unroll
            for (int i = 0; i < 4; ++i)
                af[i] = frag_a(As[cur], wm * 64 + i * 16 + lr, kk, lh);
#pragma unroll
            for (int j = 0; j < 2; ++j)
                bf[j] = frag_b(Ws[cur], wn * 32 + j * 16 + lr, kk, lh);
#pragma unroll
            for (int i = 0; i < 4; ++i)
#pragma unroll
                for (int j = 0; j < 2; ++j)
                    acc[i][j] = __builtin_amdgcn_wmma_f32_16x16x32_f16(
                        false, af[i], false, bf[j], (short)0, acc[i][j],
                        false, false);
        }
    }

    // Epilogue: C VGPR r -> (M = r + 8*lh, N = lr) within each 16x16 tile
#pragma unroll
    for (int i = 0; i < 4; ++i) {
#pragma unroll
        for (int j = 0; j < 2; ++j) {
#pragma unroll
            for (int r = 0; r < 8; ++r) {
                int m = mBase + wm * 64 + i * 16 + r + 8 * lh;
                int n = nBase + wn * 32 + j * 16 + lr;
                if (m < M && n < N) {
                    float v = acc[i][j][r];
                    if (bias) v += bias[n];
                    if (RELU) v = fmaxf(v, 0.f);
                    if (CLIP) {
                        if (__builtin_isnan(v) || __builtin_isinf(v)) v = 0.f;
                        v = fminf(fmaxf(v, -10.f), 10.f);
                    }
                    size_t off;
                    if (ROWMAP == 0)
                        off = (size_t)m * out_pitch + n;
                    else
                        off = (size_t)(m >> 1) * out_pitch +
                              (size_t)(m & 1) * 512 + n;
                    if (OUT_F16) outH[off] = (f16)v;
                    else         outF[off] = v;
                }
            }
        }
    }
}

// ---------------------------------------------------------------------------
// Per-pair attention: seq_len 2, 8 heads of 64. One wave per pair.
// ---------------------------------------------------------------------------
__global__ __launch_bounds__(256) void attn_pair_kernel(
    const float* __restrict__ qkv,     // [NROWS, 1536] f32
    const int*   __restrict__ pairs,   // [B, M, 2] int32
    f16*         __restrict__ O)       // [2*NPAIRS, 512] f16
{
    const int lane = threadIdx.x & 31;
    const int wave = threadIdx.x >> 5;
    const int p    = blockIdx.x * 8 + wave;
    if (p >= NPAIRS) return;

    const int b  = p >> 12;                 // / DIM_M (4096)
    int si = pairs[2 * p + 0];
    int oi = pairs[2 * p + 1];
    si = min(max(si, 0), DIM_N - 1);
    oi = min(max(oi, 0), DIM_N - 1);
    const int r0 = b * DIM_N + si;
    const int r1 = b * DIM_N + oi;

    const float* q0 = qkv + (size_t)r0 * DIM_3E;
    const float* q1 = qkv + (size_t)r1 * DIM_3E;
    const float* k0 = q0 + DIM_E;
    const float* k1 = q1 + DIM_E;
    const float* v0 = q0 + 2 * DIM_E;
    const float* v1 = q1 + 2 * DIM_E;

    float aw0[DIM_H];   // P(attend subj) for query position 0
    float aw1[DIM_H];   // P(attend subj) for query position 1

#pragma unroll
    for (int h = 0; h < DIM_H; ++h) {
        const int e0 = h * DIM_HD + lane;
        const int e1 = e0 + 32;
        const float qa0 = q0[e0], qb0 = q0[e1];
        const float qa1 = q1[e0], qb1 = q1[e1];
        const float ka0 = k0[e0], kb0 = k0[e1];
        const float ka1 = k1[e0], kb1 = k1[e1];
        float s00 = qa0 * ka0 + qb0 * kb0;
        float s01 = qa0 * ka1 + qb0 * kb1;
        float s10 = qa1 * ka0 + qb1 * kb0;
        float s11 = qa1 * ka1 + qb1 * kb1;
#pragma unroll
        for (int msk = 16; msk >= 1; msk >>= 1) {
            s00 += __shfl_xor(s00, msk, 32);
            s01 += __shfl_xor(s01, msk, 32);
            s10 += __shfl_xor(s10, msk, 32);
            s11 += __shfl_xor(s11, msk, 32);
        }
        s00 *= 0.125f; s01 *= 0.125f; s10 *= 0.125f; s11 *= 0.125f;
        {
            float mx = fmaxf(s00, s01);
            float ea = __expf(s00 - mx), eb = __expf(s01 - mx);
            aw0[h] = ea / (ea + eb);
        }
        {
            float mx = fmaxf(s10, s11);
            float ea = __expf(s10 - mx), eb = __expf(s11 - mx);
            aw1[h] = ea / (ea + eb);
        }
    }

    f16* out0 = O + (size_t)(2 * p) * DIM_E;
    f16* out1 = out0 + DIM_E;
#pragma unroll
    for (int h = 0; h < DIM_H; ++h) {
#pragma unroll
        for (int j = 0; j < 2; ++j) {
            const int e = h * DIM_HD + j * 32 + lane;
            const float x0 = v0[e], x1 = v1[e];
            out0[e] = (f16)(aw0[h] * x0 + (1.f - aw0[h]) * x1);
            out1[e] = (f16)(aw1[h] * x0 + (1.f - aw1[h]) * x1);
        }
    }
}

// ---------------------------------------------------------------------------
// Host orchestration
// ---------------------------------------------------------------------------
static void launch_cvt(const float* s, f16* d, int n, hipStream_t st) {
    int grid = ceil_div_i(n, 8 * 256);
    cvt_f32_f16_kernel<<<grid, 256, 0, st>>>(s, d, n);
}

template <bool RELU, bool OUT_F16, int ROWMAP, bool CLIP>
static void launch_gemm(const f16* A, const f16* W, const float* bias,
                        float* outF, f16* outH,
                        int M, int N, int K, int pitch, hipStream_t st) {
    dim3 grid(ceil_div_i(M, BM), ceil_div_i(N, BN));
    gemm_wmma_kernel<RELU, OUT_F16, ROWMAP, CLIP>
        <<<grid, 256, 0, st>>>(A, W, bias, outF, outH, M, N, K, pitch);
}

extern "C" void kernel_launch(void* const* d_in, const int* in_sizes, int n_in,
                              void* d_out, int out_size, void* d_ws, size_t ws_size,
                              hipStream_t stream) {
    (void)in_sizes; (void)n_in; (void)out_size; (void)ws_size;

    const float* visual  = (const float*)d_in[0];   // [16,100,512]
    const float* spatial = (const float*)d_in[1];   // [16,4096,512]
    const int*   pairs   = (const int*)  d_in[2];   // [16,4096,2]
    const float* w_in    = (const float*)d_in[3];   // [1536,512]
    const float* b_in    = (const float*)d_in[4];
    const float* w_out   = (const float*)d_in[5];   // [512,512]
    const float* b_out   = (const float*)d_in[6];
    const float* w_sp    = (const float*)d_in[7];   // [512,512]
    const float* b_sp    = (const float*)d_in[8];
    const float* w_fc1   = (const float*)d_in[9];   // [1024,1536]
    const float* b_fc1   = (const float*)d_in[10];
    const float* w_fc2   = (const float*)d_in[11];  // [512,1024]
    const float* b_fc2   = (const float*)d_in[12];
    const float* w_fc3   = (const float*)d_in[13];  // [50,512]
    const float* b_fc3   = (const float*)d_in[14];
    float*       out     = (float*)d_out;           // [65536,50]

    char* ws = (char*)d_ws;
    size_t off = 0;
    auto alloc = [&](size_t bytes) -> char* {
        char* p = ws + off;
        off += (bytes + 255) & ~(size_t)255;
        return p;
    };

    f16* vis16   = (f16*)alloc((size_t)NROWS * DIM_E * 2);
    f16* w_in16  = (f16*)alloc((size_t)DIM_3E * DIM_E * 2);
    f16* w_out16 = (f16*)alloc((size_t)DIM_E * DIM_E * 2);
    f16* w_sp16  = (f16*)alloc((size_t)DIM_E * DIM_E * 2);
    f16* w_fc116 = (f16*)alloc((size_t)DIM_HID * DIM_3E * 2);
    f16* w_fc216 = (f16*)alloc((size_t)DIM_H2 * DIM_HID * 2);
    f16* w_fc316 = (f16*)alloc((size_t)DIM_REL * DIM_E * 2);
    float* qkv   = (float*)alloc((size_t)NROWS * DIM_3E * 4);
    f16* sp16    = (f16*)alloc((size_t)NPAIRS * DIM_E * 2);      // also h2 alias
    f16* O16     = (f16*)alloc((size_t)2 * NPAIRS * DIM_E * 2);  // also h1 alias
    f16* comb16  = (f16*)alloc((size_t)NPAIRS * DIM_3E * 2);
    f16* h1_16   = O16;    // free after out-proj GEMM
    f16* h2_16   = sp16;   // free after spatial GEMM

    // 1) converts
    launch_cvt(visual,  vis16,   NROWS * DIM_E,    stream);
    launch_cvt(spatial, sp16,    NPAIRS * DIM_E,   stream);
    launch_cvt(w_in,    w_in16,  DIM_3E * DIM_E,   stream);
    launch_cvt(w_out,   w_out16, DIM_E * DIM_E,    stream);
    launch_cvt(w_sp,    w_sp16,  DIM_E * DIM_E,    stream);
    launch_cvt(w_fc1,   w_fc116, DIM_HID * DIM_3E, stream);
    launch_cvt(w_fc2,   w_fc216, DIM_H2 * DIM_HID, stream);
    launch_cvt(w_fc3,   w_fc316, DIM_REL * DIM_E,  stream);

    // 2) QKV on unique rows: [1600,512] @ [512,1536] + b_in -> f32
    launch_gemm<false, false, 0, false>(vis16, w_in16, b_in, qkv, nullptr,
                                        NROWS, DIM_3E, DIM_E, DIM_3E, stream);

    // 3) per-pair attention -> O f16 [131072,512]
    attn_pair_kernel<<<NPAIRS / 8, 256, 0, stream>>>(qkv, pairs, O16);

    // 4) out-proj: [131072,512] @ [512,512] + b_out -> comb cols [q*512, q*512+512)
    launch_gemm<false, true, 1, false>(O16, w_out16, b_out, nullptr, comb16,
                                       2 * NPAIRS, DIM_E, DIM_E, DIM_3E, stream);

    // 5) spatial: [65536,512] @ [512,512] + b_sp -> comb cols [1024,1536)
    launch_gemm<false, true, 0, false>(sp16, w_sp16, b_sp, nullptr,
                                       comb16 + 2 * DIM_E,
                                       NPAIRS, DIM_E, DIM_E, DIM_3E, stream);

    // 6) fc1 + relu: [65536,1536] @ [1536,1024]
    launch_gemm<true, true, 0, false>(comb16, w_fc116, b_fc1, nullptr, h1_16,
                                      NPAIRS, DIM_HID, DIM_3E, DIM_HID, stream);

    // 7) fc2 + relu: [65536,1024] @ [1024,512]
    launch_gemm<true, true, 0, false>(h1_16, w_fc216, b_fc2, nullptr, h2_16,
                                      NPAIRS, DIM_H2, DIM_HID, DIM_H2, stream);

    // 8) fc3 + bias + nan-scrub + clip(+-10): [65536,512] @ [512,50] -> d_out
    launch_gemm<false, false, 0, true>(h2_16, w_fc316, b_fc3, out, nullptr,
                                       NPAIRS, DIM_REL, DIM_E, DIM_REL, stream);
}
